// HotelPredModel_68410239091340
// MI455X (gfx1250) — matrix operand order repaired and checked by reading, not compiled, not run on hardware
//
#include <hip/hip_runtime.h>

typedef __attribute__((ext_vector_type(16))) _Float16 v16h;
typedef __attribute__((ext_vector_type(8)))  _Float16 v8h;
typedef __attribute__((ext_vector_type(4)))  _Float16 v4h;
typedef __attribute__((ext_vector_type(8)))  float    v8f;

#define TT  512   // timesteps
#define FIN 32    // input features (layer 0)
#define HID 64    // hidden size
#define NG  256   // 4*HID gate columns
#define BT  16    // batch rows per workgroup (WMMA M)

__device__ __forceinline__ v8f wmma16(v16h a, v16h b, v8f c) {
  // D = A(16x32 f16) * B(32x16 f16) + C(16x16 f32)
  return __builtin_amdgcn_wmma_f32_16x16x32_f16(false, a, false, b, (short)0, c,
                                                false, false);
}

// A-fragment from f16 row-major LDS tile (row stride `stride` halves).
// 16-bit A layout: lanes 0-15 row m=lane, K = cbase+0..7 (v0..3) and
// cbase+16..23 (v4..7); lanes 16-31 same rows, K offset +8.
__device__ __forceinline__ v16h ldA(const _Float16* p, int stride, int lane, int cbase) {
  const int m  = lane & 15;
  const int kb = cbase + ((lane & 16) ? 8 : 0);
  v8h lo = *(const v8h*)(p + m * stride + kb);
  v8h hi = *(const v8h*)(p + m * stride + kb + 16);
  return __builtin_shufflevector(lo, hi, 0, 1, 2, 3, 4, 5, 6, 7,
                                 8, 9, 10, 11, 12, 13, 14, 15);
}

// B-fragment: B[k][n] = W[n][k] (W row-major, ncol K-columns).
// lane = column n (lanes 16-31: K offset +8); K = k0+0..7 and k0+16..23 per lane.
__device__ __forceinline__ v16h ldB(const float* W, int ncol, int n, int k0) {
  const float* p = W + (size_t)n * ncol + k0;
  v16h r;
#pragma unroll
  for (int i = 0; i < 16; ++i) r[i] = (_Float16)p[(i & 7) + ((i & 8) ? 16 : 0)];
  return r;
}

__device__ __forceinline__ float fast_rcp(float x) {
#if __has_builtin(__builtin_amdgcn_rcpf)
  return __builtin_amdgcn_rcpf(x);   // v_rcp_f32 (TRANS)
#else
  return 1.0f / x;
#endif
}

__device__ __forceinline__ float tanh_fast(float x) {
#if __has_builtin(__builtin_amdgcn_tanhf)
  return __builtin_amdgcn_tanhf(x);  // native v_tanh_f32 on gfx1250
#else
  return 2.0f * fast_rcp(1.0f + __expf(-2.0f * x)) - 1.0f;
#endif
}

__device__ __forceinline__ float sigm(float x) {
#if __has_builtin(__builtin_amdgcn_tanhf)
  // sigmoid(x) = 0.5*tanh(0.5x) + 0.5 : single TRANS op + mul + fma
  return __builtin_fmaf(0.5f, __builtin_amdgcn_tanhf(0.5f * x), 0.5f);
#else
  return fast_rcp(1.0f + __expf(-x));
#endif
}

__global__ __launch_bounds__(256) void lstm2_fused_kernel(
    const float* __restrict__ x,
    const float* __restrict__ Wih0, const float* __restrict__ Whh0,
    const float* __restrict__ bias0,
    const float* __restrict__ Wih1, const float* __restrict__ Whh1,
    const float* __restrict__ bias1,
    const float* __restrict__ Wfc, const float* __restrict__ bfc,
    float* __restrict__ out) {
  __shared__ __align__(16) _Float16 xls[BT * FIN];    // x_t tile (f16)
  __shared__ __align__(16) _Float16 h0ls[BT * HID];   // layer-0 hidden (f16)
  __shared__ __align__(16) _Float16 h1ls[BT * HID];   // layer-1 hidden (f16)
  __shared__ __align__(16) float    gls[BT * NG];     // gate pre-activations
  __shared__ __align__(16) float    hout[BT * HID];   // final h1 (f32)

  const int tid  = threadIdx.x;
  const int lane = tid & 31;
  const int wv   = tid >> 5;           // wave 0..7, owns gate tiles 2wv, 2wv+1
  const int brow = blockIdx.x * BT;

  const int nlo   = lane & 15;
  const int khalf = (lane & 16) ? 8 : 0;

  // ---------- preload all weight B-fragments into VGPRs (time-invariant) ----
  v16h wih0[2], whh0[2][2], wih1[2][2], whh1[2][2];
  float bv0[2], bv1[2];
#pragma unroll
  for (int tt = 0; tt < 2; ++tt) {
    const int n = (2 * wv + tt) * 16 + nlo;   // gate column owned by this lane
    wih0[tt]    = ldB(Wih0, FIN, n, khalf);          // K=32: one chunk
    whh0[tt][0] = ldB(Whh0, HID, n, 0 + khalf);      // K=64: two chunks
    whh0[tt][1] = ldB(Whh0, HID, n, 32 + khalf);
    wih1[tt][0] = ldB(Wih1, HID, n, 0 + khalf);
    wih1[tt][1] = ldB(Wih1, HID, n, 32 + khalf);
    whh1[tt][0] = ldB(Whh1, HID, n, 0 + khalf);
    whh1[tt][1] = ldB(Whh1, HID, n, 32 + khalf);
    bv0[tt] = bias0[n];
    bv1[tt] = bias1[n];
  }

  // ---------- zero initial hidden/cell state ----------
  {
    v4h z;
    z[0] = z[1] = z[2] = z[3] = (_Float16)0.0f;
    *(v4h*)&h0ls[tid * 4] = z;   // 256 threads * 4 = 1024 halves
    *(v4h*)&h1ls[tid * 4] = z;
  }
  float c0[4] = {0.f, 0.f, 0.f, 0.f};
  float c1[4] = {0.f, 0.f, 0.f, 0.f};
  const int em = tid >> 4;          // elementwise: batch row
  const int ej = (tid & 15) << 2;   // elementwise: 4 hidden units
  __syncthreads();

  for (int t = 0; t < TT; ++t) {
    // ---- stage x_t (16 x 32 f32 -> f16 LDS) ----
    {
      int e = tid, m = e >> 5, f = e & 31;
      xls[e] = (_Float16)x[((size_t)(brow + m) * TT + t) * FIN + f];
      e = tid + 256; m = e >> 5; f = e & 31;
      xls[e] = (_Float16)x[((size_t)(brow + m) * TT + t) * FIN + f];
    }
    __syncthreads();

    // ---- layer 0 GEMMs: g = b0 + x_t @ Wih0^T + h0_{t-1} @ Whh0^T ----
    {
      v16h ax  = ldA(xls, FIN, lane, 0);
      v16h ah0 = ldA(h0ls, HID, lane, 0);
      v16h ah1 = ldA(h0ls, HID, lane, 32);
#pragma unroll
      for (int tt = 0; tt < 2; ++tt) {
        v8f acc;
#pragma unroll
        for (int r = 0; r < 8; ++r) acc[r] = bv0[tt];
        acc = wmma16(ax, wih0[tt], acc);
        acc = wmma16(ah0, whh0[tt][0], acc);
        acc = wmma16(ah1, whh0[tt][1], acc);
        const int nb = (2 * wv + tt) * 16 + nlo;
        const int mb = (lane & 16) ? 8 : 0;   // C layout: vgpr r -> row mb+r
#pragma unroll
        for (int r = 0; r < 8; ++r) gls[(mb + r) * NG + nb] = acc[r];
      }
    }
    __syncthreads();

    // ---- layer 0 elementwise: c,h update ----
    {
      const float* g = &gls[em * NG + ej];
      v4h hv;
#pragma unroll
      for (int q = 0; q < 4; ++q) {
        float iv = sigm(g[q]);
        float fv = sigm(g[64 + q]);
        float gv = tanh_fast(g[128 + q]);
        float ov = sigm(g[192 + q]);
        float c  = fv * c0[q] + iv * gv;
        c0[q]    = c;
        hv[q]    = (_Float16)(ov * tanh_fast(c));
      }
      *(v4h*)&h0ls[em * HID + ej] = hv;
    }
    __syncthreads();

    // ---- layer 1 GEMMs: g = b1 + h0_t @ Wih1^T + h1_{t-1} @ Whh1^T ----
    {
      v16h ai0 = ldA(h0ls, HID, lane, 0);
      v16h ai1 = ldA(h0ls, HID, lane, 32);
      v16h ar0 = ldA(h1ls, HID, lane, 0);
      v16h ar1 = ldA(h1ls, HID, lane, 32);
#pragma unroll
      for (int tt = 0; tt < 2; ++tt) {
        v8f acc;
#pragma unroll
        for (int r = 0; r < 8; ++r) acc[r] = bv1[tt];
        acc = wmma16(ai0, wih1[tt][0], acc);
        acc = wmma16(ai1, wih1[tt][1], acc);
        acc = wmma16(ar0, whh1[tt][0], acc);
        acc = wmma16(ar1, whh1[tt][1], acc);
        const int nb = (2 * wv + tt) * 16 + nlo;
        const int mb = (lane & 16) ? 8 : 0;
#pragma unroll
        for (int r = 0; r < 8; ++r) gls[(mb + r) * NG + nb] = acc[r];
      }
    }
    __syncthreads();

    // ---- layer 1 elementwise ----
    {
      const float* g = &gls[em * NG + ej];
      v4h hv;
#pragma unroll
      for (int q = 0; q < 4; ++q) {
        float iv = sigm(g[q]);
        float fv = sigm(g[64 + q]);
        float gv = tanh_fast(g[128 + q]);
        float ov = sigm(g[192 + q]);
        float c  = fv * c1[q] + iv * gv;
        c1[q]    = c;
        float h  = ov * tanh_fast(c);
        hv[q]    = (_Float16)h;
        if (t == TT - 1) hout[em * HID + ej + q] = h;  // uniform branch
      }
      *(v4h*)&h1ls[em * HID + ej] = hv;
    }
    __syncthreads();
  }

  // ---- final FC: out[b] = h1_last[b,:] . Wfc + bfc ----
  if (tid < BT) {
    float s = bfc[0];
#pragma unroll 8
    for (int j = 0; j < HID; ++j) s += hout[tid * HID + j] * Wfc[j];
    out[brow + tid] = s;
  }
}

extern "C" void kernel_launch(void* const* d_in, const int* in_sizes, int n_in,
                              void* d_out, int out_size, void* d_ws, size_t ws_size,
                              hipStream_t stream) {
  const float* x    = (const float*)d_in[0];
  const float* Wih0 = (const float*)d_in[1];
  const float* Whh0 = (const float*)d_in[2];
  const float* b0   = (const float*)d_in[3];
  const float* Wih1 = (const float*)d_in[4];
  const float* Whh1 = (const float*)d_in[5];
  const float* b1   = (const float*)d_in[6];
  const float* Wfc  = (const float*)d_in[7];
  const float* bfc  = (const float*)d_in[8];
  float* out        = (float*)d_out;

  const int B = in_sizes[0] / (TT * FIN);  // 2048
  lstm2_fused_kernel<<<B / BT, 256, 0, stream>>>(x, Wih0, Whh0, b0, Wih1, Whh1,
                                                 b1, Wfc, bfc, out);
}
